// VN_Resnet_Encoder_66391604461763
// MI455X (gfx1250) — matrix operand order repaired and checked by reading, not compiled, not run on hardware
//
#include <hip/hip_runtime.h>
#include <hip/hip_bf16.h>

#define BB   8
#define NN   1024
#define KK   20
#define HH   128
#define MM   (BB * 3 * NN)   // 24576 spatial positions (b, vec-dim, n)
#define EPSV 1e-6f

typedef __attribute__((ext_vector_type(16))) __bf16 v16bf;
typedef __attribute__((ext_vector_type(8)))  float  v8f;

// ---------------------------------------------------------------------------
// fp32 -> bf16 conversion (weights)
// ---------------------------------------------------------------------------
__global__ void f2bf_kernel(const float* __restrict__ src, __bf16* __restrict__ dst, int n) {
    int i = blockIdx.x * 256 + threadIdx.x;
    if (i < n) dst[i] = (__bf16)src[i];
}

// ---------------------------------------------------------------------------
// kNN: per (b,i) find top-20 nearest (max of negative squared distance),
// stable tie-break toward lower index (matches jax.lax.top_k).
// Points of one batch staged in LDS.
// ---------------------------------------------------------------------------
__global__ void knn_kernel(const float* __restrict__ x, int* __restrict__ idx) {
    __shared__ float px[NN], py[NN], pz[NN];
    int b = blockIdx.x >> 2;
    int i = ((blockIdx.x & 3) << 8) + threadIdx.x;
    const float* gx = x + (size_t)b * NN * 3;
    for (int t = threadIdx.x; t < NN; t += 256) {
        px[t] = gx[t * 3 + 0];
        py[t] = gx[t * 3 + 1];
        pz[t] = gx[t * 3 + 2];
    }
    __syncthreads();
    float vx = px[i], vy = py[i], vz = pz[i];
    float val[KK];
    int   id[KK];
    #pragma unroll
    for (int t = 0; t < KK; ++t) { val[t] = -3.4e38f; id[t] = 0; }
    for (int j = 0; j < NN; ++j) {
        float dx = vx - px[j], dy = vy - py[j], dz = vz - pz[j];
        float v = -(dx * dx + dy * dy + dz * dz);
        if (v <= val[KK - 1]) continue;
        int p = KK - 1;
        while (p > 0 && v > val[p - 1]) {
            val[p] = val[p - 1]; id[p] = id[p - 1]; --p;
        }
        val[p] = v; id[p] = j;
    }
    int* out = idx + ((size_t)b * NN + i) * KK;
    #pragma unroll
    for (int t = 0; t < KK; ++t) out[t] = id[t];
}

// ---------------------------------------------------------------------------
// Edge features + conv_pos (C=3 -> H=128) + VN-leakyReLU(0.2) + mean over k.
// One block of 128 threads per (b,n). Writes Xh0 bf16 [M][128].
// feat channels: c0 = neigh-center, c1 = center, c2 = cross(neigh, center)
// ---------------------------------------------------------------------------
__global__ void feat_conv_kernel(const float* __restrict__ x, const int* __restrict__ idx,
                                 const float* __restrict__ Wf, const float* __restrict__ Wd,
                                 __bf16* __restrict__ Xh0) {
    __shared__ float sfeat[3][3][KK];     // [channel][dim][k]
    __shared__ float sWf[HH * 3], sWd[HH * 3];
    int bn = blockIdx.x;
    int b = bn >> 10, n = bn & 1023;
    int t = threadIdx.x;                  // 0..127
    for (int q = t; q < HH * 3; q += 128) { sWf[q] = Wf[q]; sWd[q] = Wd[q]; }
    const float* gx = x + (size_t)b * NN * 3;
    float cx = gx[n * 3 + 0], cy = gx[n * 3 + 1], cz = gx[n * 3 + 2];
    if (t < KK) {
        int j = idx[((size_t)b * NN + n) * KK + t];
        float nx = gx[j * 3 + 0], ny = gx[j * 3 + 1], nz = gx[j * 3 + 2];
        sfeat[0][0][t] = nx - cx; sfeat[0][1][t] = ny - cy; sfeat[0][2][t] = nz - cz;
        sfeat[1][0][t] = cx;      sfeat[1][1][t] = cy;      sfeat[1][2][t] = cz;
        sfeat[2][0][t] = ny * cz - nz * cy;   // cross(neigh, center)
        sfeat[2][1][t] = nz * cx - nx * cz;
        sfeat[2][2][t] = nx * cy - ny * cx;
    }
    __syncthreads();
    int o = t;
    float wf0 = sWf[o * 3], wf1 = sWf[o * 3 + 1], wf2 = sWf[o * 3 + 2];
    float wd0 = sWd[o * 3], wd1 = sWd[o * 3 + 1], wd2 = sWd[o * 3 + 2];
    float acc0 = 0.f, acc1 = 0.f, acc2 = 0.f;
    for (int kk = 0; kk < KK; ++kk) {
        float p0 = wf0 * sfeat[0][0][kk] + wf1 * sfeat[1][0][kk] + wf2 * sfeat[2][0][kk];
        float p1 = wf0 * sfeat[0][1][kk] + wf1 * sfeat[1][1][kk] + wf2 * sfeat[2][1][kk];
        float p2 = wf0 * sfeat[0][2][kk] + wf1 * sfeat[1][2][kk] + wf2 * sfeat[2][2][kk];
        float q0 = wd0 * sfeat[0][0][kk] + wd1 * sfeat[1][0][kk] + wd2 * sfeat[2][0][kk];
        float q1 = wd0 * sfeat[0][1][kk] + wd1 * sfeat[1][1][kk] + wd2 * sfeat[2][1][kk];
        float q2 = wd0 * sfeat[0][2][kk] + wd1 * sfeat[1][2][kk] + wd2 * sfeat[2][2][kk];
        float dot = p0 * q0 + p1 * q1 + p2 * q2;
        float dsq = q0 * q0 + q1 * q1 + q2 * q2;
        float s = dot / (dsq + EPSV);
        float keep = (dot >= 0.f) ? 1.f : 0.f;
        acc0 += 0.2f * p0 + 0.8f * (keep * p0 + (1.f - keep) * (p0 - s * q0));
        acc1 += 0.2f * p1 + 0.8f * (keep * p1 + (1.f - keep) * (p1 - s * q1));
        acc2 += 0.2f * p2 + 0.8f * (keep * p2 + (1.f - keep) * (p2 - s * q2));
    }
    float inv_k = 1.f / (float)KK;
    int m0 = (b * 3 + 0) * NN + n;
    int m1 = (b * 3 + 1) * NN + n;
    int m2 = (b * 3 + 2) * NN + n;
    Xh0[(size_t)m0 * HH + o] = (__bf16)(acc0 * inv_k);
    Xh0[(size_t)m1 * HH + o] = (__bf16)(acc1 * inv_k);
    Xh0[(size_t)m2 * HH + o] = (__bf16)(acc2 * inv_k);
}

// ---------------------------------------------------------------------------
// WMMA GEMM, register-blocked 64(O) x 16(M) per wave:
//   Y[o][m] (f32 [O][M])  =  W[o][c] (bf16 [O][C]) @ X[m][c] (bf16 [M][C])
// B-fragment (activations) is reused by 4 WMMAs per K-step.
// ACCUM:  Y += W@X  (folds the residual add into the shortcut GEMM)
// Xout :  optional bf16 [M][O] transposed copy of the result (next GEMM input)
// A/B fragments use contiguous k = 16*(lane/16) + element, same in both (wave32).
// ---------------------------------------------------------------------------
template <int C, bool ACCUM>
__global__ __launch_bounds__(256)
void gemm_wmma_kernel(const __bf16* __restrict__ W, const __bf16* __restrict__ X,
                      float* __restrict__ Y, __bf16* __restrict__ Xout,
                      int O, int M) {
    int wave  = threadIdx.x >> 5;
    int lane  = threadIdx.x & 31;
    int mtile = blockIdx.x * 8 + wave;     // 16-col tile of M
    int ob    = blockIdx.y * 64;           // 64-row block of O
    int mb    = mtile * 16;
    int r     = lane & 15;
    int khalf = lane >> 4;

    const __bf16* wrow[4];
    #pragma unroll
    for (int t = 0; t < 4; ++t) wrow[t] = W + (size_t)(ob + t * 16 + r) * C;
    const __bf16* xrow = X + (size_t)(mb + r) * C + (khalf << 4);
    __builtin_prefetch(wrow[0], 0, 3);     // global_prefetch_b8
    __builtin_prefetch(xrow, 0, 3);

    v8f acc[4];
    #pragma unroll
    for (int t = 0; t < 4; ++t) acc[t] = v8f{0.f, 0.f, 0.f, 0.f, 0.f, 0.f, 0.f, 0.f};

    #pragma unroll
    for (int c0 = 0; c0 < C; c0 += 32) {
        // B: 32x16 bf16. col = lane%16; lanes 0-15 K 0..15, lanes 16-31 K 16..31
        v16bf bfrag;
        #pragma unroll
        for (int j = 0; j < 8; ++j) {
            bfrag[2 * j]     = xrow[c0 + 2 * j];
            bfrag[2 * j + 1] = xrow[c0 + 2 * j + 1];
        }
        // A: 16x32 bf16. lane<16 holds K 0..15 +c0, lane>=16 K 16..31 +c0
        #pragma unroll
        for (int t = 0; t < 4; ++t) {
            v16bf afrag;
            #pragma unroll
            for (int j = 0; j < 8; ++j) {
                int k0 = (khalf << 4) + (j << 1);
                afrag[2 * j]     = wrow[t][c0 + k0];
                afrag[2 * j + 1] = wrow[t][c0 + k0 + 1];
            }
            acc[t] = __builtin_amdgcn_wmma_f32_16x16x32_bf16(
                false, afrag, false, bfrag, (short)0, acc[t], false, false);
        }
    }
    // D: col m = mb + lane%16; VGPR j -> row o = ob' + j (+8 for upper lanes)
    #pragma unroll
    for (int t = 0; t < 4; ++t) {
        #pragma unroll
        for (int j = 0; j < 8; ++j) {
            int o = ob + t * 16 + (khalf ? j + 8 : j);
            size_t yi = (size_t)o * M + (mb + r);
            float v = acc[t][j];
            if (ACCUM) v += Y[yi];
            Y[yi] = v;
            if (Xout != nullptr) Xout[(size_t)(mb + r) * O + o] = (__bf16)v;
        }
    }
}

// ---------------------------------------------------------------------------
// VN leaky-ReLU: p from bf16 X [M][O], direction from f32 D [O][M],
// result bf16 A [M][O] (transposed layout ready as next GEMM B operand).
// ---------------------------------------------------------------------------
__global__ void vn_act_kernel(const __bf16* __restrict__ Xp, const float* __restrict__ D,
                              __bf16* __restrict__ A, int O, float slope) {
    int tid = blockIdx.x * 256 + threadIdx.x;
    int total = BB * NN * O;
    if (tid >= total) return;
    int o = tid % O;
    int n = (tid / O) % NN;
    int b = tid / (O * NN);
    float p[3], q[3];
    int m[3];
    #pragma unroll
    for (int d = 0; d < 3; ++d) {
        m[d] = (b * 3 + d) * NN + n;
        p[d] = (float)Xp[(size_t)m[d] * O + o];
        q[d] = D[(size_t)o * MM + m[d]];
    }
    float dot = p[0] * q[0] + p[1] * q[1] + p[2] * q[2];
    float dsq = q[0] * q[0] + q[1] * q[1] + q[2] * q[2];
    float s = dot / (dsq + EPSV);
    float keep = (dot >= 0.f) ? 1.f : 0.f;
    #pragma unroll
    for (int d = 0; d < 3; ++d) {
        float pn  = p[d] - s * q[d];
        float out = slope * p[d] + (1.f - slope) * (keep * p[d] + (1.f - keep) * pn);
        A[(size_t)m[d] * O + o] = (__bf16)out;
    }
}

// mean over n: Pool[(b*H+o)*3+d] = mean_n H[o][(b*3+d)*N+n]
__global__ void pool_kernel(const float* __restrict__ H, float* __restrict__ Pool) {
    int tid = blockIdx.x * 256 + threadIdx.x;
    if (tid >= BB * HH * 3) return;
    int d = tid % 3;
    int o = (tid / 3) % HH;
    int b = tid / (3 * HH);
    const float* row = H + (size_t)o * MM + (size_t)(b * 3 + d) * NN;
    float s = 0.f;
    for (int n = 0; n < NN; ++n) s += row[n];
    Pool[tid] = s / (float)NN;
}

// next hidden input: X[m][c] = c<H ? H[c][m] : Pool[b][c-H][d]   (bf16, C=256)
__global__ void build_next_kernel(const float* __restrict__ H, const float* __restrict__ Pool,
                                  __bf16* __restrict__ X) {
    int tid = blockIdx.x * 256 + threadIdx.x;
    int total = MM * 256;
    if (tid >= total) return;
    int c = tid % 256;
    int m = tid / 256;
    int b = m / (3 * NN);
    int d = (m / NN) % 3;
    float v = (c < HH) ? H[(size_t)c * MM + m] : Pool[(b * HH + (c - HH)) * 3 + d];
    X[(size_t)m * 256 + c] = (__bf16)v;
}

// head: act = vn_lrelu(pool, actvn_c_Wd @ pool, 0.2)
__global__ void head_act_kernel(const float* __restrict__ Pool, const float* __restrict__ Wd,
                                float* __restrict__ Act) {
    int tid = blockIdx.x * 256 + threadIdx.x;
    if (tid >= BB * HH) return;
    int o = tid % HH;
    int b = tid / HH;
    float q0 = 0.f, q1 = 0.f, q2 = 0.f;
    for (int c = 0; c < HH; ++c) {
        float w = Wd[o * HH + c];
        const float* pc = Pool + (b * HH + c) * 3;
        q0 += w * pc[0]; q1 += w * pc[1]; q2 += w * pc[2];
    }
    const float* pp = Pool + (b * HH + o) * 3;
    float p0 = pp[0], p1 = pp[1], p2 = pp[2];
    float dot = p0 * q0 + p1 * q1 + p2 * q2;
    float dsq = q0 * q0 + q1 * q1 + q2 * q2;
    float s = dot / (dsq + EPSV);
    float keep = (dot >= 0.f) ? 1.f : 0.f;
    float* ao = Act + (b * HH + o) * 3;
    ao[0] = 0.2f * p0 + 0.8f * (keep * p0 + (1.f - keep) * (p0 - s * q0));
    ao[1] = 0.2f * p1 + 0.8f * (keep * p1 + (1.f - keep) * (p1 - s * q1));
    ao[2] = 0.2f * p2 + 0.8f * (keep * p2 + (1.f - keep) * (p2 - s * q2));
}

// head: out[b][l][d] = sum_c fc_c_W[l][c] * Act[b][c][d]
__global__ void head_out_kernel(const float* __restrict__ Act, const float* __restrict__ Wc,
                                float* __restrict__ out) {
    int tid = blockIdx.x * 256 + threadIdx.x;
    if (tid >= BB * 128) return;
    int l = tid % 128;
    int b = tid / 128;
    float s0 = 0.f, s1 = 0.f, s2 = 0.f;
    for (int c = 0; c < HH; ++c) {
        float w = Wc[l * HH + c];
        const float* ac = Act + (b * HH + c) * 3;
        s0 += w * ac[0]; s1 += w * ac[1]; s2 += w * ac[2];
    }
    float* op = out + (size_t)b * 128 * 3 + l * 3;
    op[0] = s0; op[1] = s1; op[2] = s2;
}

// ---------------------------------------------------------------------------
extern "C" void kernel_launch(void* const* d_in, const int* in_sizes, int n_in,
                              void* d_out, int out_size, void* d_ws, size_t ws_size,
                              hipStream_t stream) {
    const float* x      = (const float*)d_in[0];   // (B,1,N,3)
    const float* cWf    = (const float*)d_in[1];   // (128,3)
    const float* cWd    = (const float*)d_in[2];   // (128,3)
    const float* fc_pos = (const float*)d_in[3];   // (256,128)
    const float* Wd0    = (const float*)d_in[4];   // (4,256,256)
    const float* W0     = (const float*)d_in[5];   // (4,128,256)
    const float* Wd1    = (const float*)d_in[6];   // (4,128,128)
    const float* W1     = (const float*)d_in[7];   // (4,128,128)
    const float* Ws     = (const float*)d_in[8];   // (4,128,256)
    const float* aWd    = (const float*)d_in[9];   // (128,128)
    const float* fcc    = (const float*)d_in[10];  // (128,128)

    // --- carve workspace (all offsets 256B aligned) ---
    char* ws = (char*)d_ws;
    auto carve = [&](size_t bytes) -> void* {
        void* p = (void*)ws;
        ws += (bytes + 255) & ~(size_t)255;
        return p;
    };
    int*    idx   = (int*)   carve((size_t)BB * NN * KK * 4);
    __bf16* bFc   = (__bf16*)carve((size_t)256 * 128 * 2);
    __bf16* bWd0  = (__bf16*)carve((size_t)4 * 256 * 256 * 2);
    __bf16* bW0   = (__bf16*)carve((size_t)4 * 128 * 256 * 2);
    __bf16* bWd1  = (__bf16*)carve((size_t)4 * 128 * 128 * 2);
    __bf16* bW1   = (__bf16*)carve((size_t)4 * 128 * 128 * 2);
    __bf16* bWs   = (__bf16*)carve((size_t)4 * 128 * 256 * 2);
    __bf16* Xhid  = (__bf16*)carve((size_t)MM * 256 * 2);   // hidden, bf16 [M][256]
    __bf16* A0    = (__bf16*)carve((size_t)MM * 256 * 2);
    __bf16* X128  = (__bf16*)carve((size_t)MM * 128 * 2);   // h0 / net, bf16 [M][128]
    __bf16* A1    = (__bf16*)carve((size_t)MM * 128 * 2);
    float*  Ybig  = (float*) carve((size_t)MM * 256 * 4);   // [256][M]
    float*  Y128  = (float*) carve((size_t)MM * 128 * 4);   // [128][M]
    float*  Pool  = (float*) carve((size_t)BB * HH * 3 * 4);
    float*  Act   = (float*) carve((size_t)BB * HH * 3 * 4);

    // --- weight conversion to bf16 ---
    f2bf_kernel<<<(256 * 128 + 255) / 256, 256, 0, stream>>>(fc_pos, bFc, 256 * 128);
    f2bf_kernel<<<(4 * 256 * 256 + 255) / 256, 256, 0, stream>>>(Wd0, bWd0, 4 * 256 * 256);
    f2bf_kernel<<<(4 * 128 * 256 + 255) / 256, 256, 0, stream>>>(W0, bW0, 4 * 128 * 256);
    f2bf_kernel<<<(4 * 128 * 128 + 255) / 256, 256, 0, stream>>>(Wd1, bWd1, 4 * 128 * 128);
    f2bf_kernel<<<(4 * 128 * 128 + 255) / 256, 256, 0, stream>>>(W1, bW1, 4 * 128 * 128);
    f2bf_kernel<<<(4 * 128 * 256 + 255) / 256, 256, 0, stream>>>(Ws, bWs, 4 * 128 * 256);

    // --- kNN + edge features + conv_pos + mean_k -> X128 bf16 [M][128] ---
    knn_kernel<<<BB * (NN / 256), 256, 0, stream>>>(x, idx);
    feat_conv_kernel<<<BB * NN, 128, 0, stream>>>(x, idx, cWf, cWd, X128);

    dim3 gO256(MM / 128, 256 / 64);
    dim3 gO128(MM / 128, 128 / 64);

    // --- fc_pos: hidden = fc_pos_W @ h0 ; also emit bf16 [M][256] ---
    gemm_wmma_kernel<128, false><<<gO256, 256, 0, stream>>>(bFc, X128, Ybig, Xhid, 256, MM);

    for (int i = 0; i < 4; ++i) {
        // d0 = Wd0 @ hidden ; a0 = vn_relu(hidden, d0)
        gemm_wmma_kernel<256, false><<<gO256, 256, 0, stream>>>(
            bWd0 + (size_t)i * 256 * 256, Xhid, Ybig, (__bf16*)nullptr, 256, MM);
        vn_act_kernel<<<(BB * NN * 256 + 255) / 256, 256, 0, stream>>>(Xhid, Ybig, A0, 256, 0.f);
        // net = W0 @ a0 (also emit bf16 [M][128])
        gemm_wmma_kernel<256, false><<<gO128, 256, 0, stream>>>(
            bW0 + (size_t)i * 128 * 256, A0, Y128, X128, 128, MM);
        // d1 = Wd1 @ net ; a1 = vn_relu(net, d1)
        gemm_wmma_kernel<128, false><<<gO128, 256, 0, stream>>>(
            bWd1 + (size_t)i * 128 * 128, X128, Y128, (__bf16*)nullptr, 128, MM);
        vn_act_kernel<<<(BB * NN * 128 + 255) / 256, 256, 0, stream>>>(X128, Y128, A1, 128, 0.f);
        // h = W1 @ a1 + Ws @ hidden   (shortcut GEMM accumulates in place)
        gemm_wmma_kernel<128, false><<<gO128, 256, 0, stream>>>(
            bW1 + (size_t)i * 128 * 128, A1, Y128, (__bf16*)nullptr, 128, MM);
        gemm_wmma_kernel<256, true><<<gO128, 256, 0, stream>>>(
            bWs + (size_t)i * 128 * 256, Xhid, Y128, (__bf16*)nullptr, 128, MM);
        // pooled mean over n
        pool_kernel<<<(BB * HH * 3 + 255) / 256, 256, 0, stream>>>(Y128, Pool);
        if (i < 3)
            build_next_kernel<<<(MM * 256 + 255) / 256, 256, 0, stream>>>(Y128, Pool, Xhid);
    }

    // --- head: final hidden == Pool (B,128,3) ---
    head_act_kernel<<<(BB * HH + 255) / 256, 256, 0, stream>>>(Pool, aWd, Act);
    head_out_kernel<<<(BB * 128 + 255) / 256, 256, 0, stream>>>(Act, fcc, (float*)d_out);
}
